// AnswerModule_13469017440371
// MI455X (gfx1250) — matrix operand order repaired and checked by reading, not compiled, not run on hardware
//
#include <hip/hip_runtime.h>

// ---------------------------------------------------------------------------
// Problem constants (match reference)
// ---------------------------------------------------------------------------
#define B_   128
#define L_   512
#define H2_  1024
#define N3_  3072   // 3*H2
#define STEP_ 8
// 6-of-8 step subset (reference's INCLUDED from np rng). Mask bit i == step i
// included. Mean scale = 1/popcount(mask). Edit mask if exact subset differs.
#define INCLUDED_MASK 0xDB   // steps {0,1,3,4,6,7}
#define INCLUDED_CNT  6

typedef _Float16       v8h  __attribute__((ext_vector_type(8)));
typedef _Float16       v16h __attribute__((ext_vector_type(16)));
typedef float          v8f  __attribute__((ext_vector_type(8)));
typedef float          v4f  __attribute__((ext_vector_type(4)));
typedef unsigned short v8u  __attribute__((ext_vector_type(8)));
typedef unsigned short v4u  __attribute__((ext_vector_type(4)));

__device__ __forceinline__ float bf2f(unsigned short u) {
  return __uint_as_float(((unsigned)u) << 16);
}
__device__ __forceinline__ unsigned short f2bf(float f) {
  unsigned u = __float_as_uint(f);
  return (unsigned short)((u + 0x7FFFu + ((u >> 16) & 1u)) >> 16); // RNE
}

// ---------------------------------------------------------------------------
// One-time conversions
// ---------------------------------------------------------------------------
__global__ void k_f32_to_f16(const float* __restrict__ in,
                             _Float16* __restrict__ out, int n) {
  int i = blockIdx.x * blockDim.x + threadIdx.x;
  if (i < n) out[i] = (_Float16)in[i];
}

// out[n*H2 + k] = (f16) in[k*H2 + n]   (theta_3 transposed -> B^T, K-contig)
__global__ void k_transpose_f16(const float* __restrict__ in,
                                _Float16* __restrict__ out) {
  int i = blockIdx.x * blockDim.x + threadIdx.x; // i = n*H2 + k
  int n = i >> 10, k = i & (H2_ - 1);
  out[i] = (_Float16)in[k * H2_ + n];
}

// m_p fp32 -> bf16 (4 elts/thread, vectorized)
__global__ void k_f32_to_bf16x4(const float* __restrict__ in,
                                unsigned short* __restrict__ out, int n4) {
  int i = blockIdx.x * blockDim.x + threadIdx.x;
  if (i >= n4) return;
  v4f v = *(const v4f*)(in + (size_t)i * 4);
  v4u r;
#pragma unroll
  for (int j = 0; j < 4; ++j) r[j] = f2bf(v[j]);
  *(v4u*)(out + (size_t)i * 4) = r;
}

__global__ void k_zero(float* __restrict__ p, int n) {
  int i = blockIdx.x * blockDim.x + threadIdx.x;
  if (i < n) p[i] = 0.f;
}

// ---------------------------------------------------------------------------
// Attention logits: logits[b,l] = q_b . M[b,l,:]   (one wave per (b,l))
// use16 -> M is bf16 (preferred, L2-resident), else fp32.
// ---------------------------------------------------------------------------
__global__ void k_attn_dots(const unsigned short* __restrict__ m16,
                            const float* __restrict__ m32, int use16,
                            const float* __restrict__ q, int qStride,
                            float* __restrict__ logits) {
  int gid  = blockIdx.x * blockDim.x + threadIdx.x;
  int wave = gid >> 5, lane = gid & 31;
  int b = wave >> 9;          // / L_
  int l = wave & (L_ - 1);
  const float* qp = q + (size_t)b * qStride;
  size_t row = ((size_t)b * L_ + l) * H2_;
  float acc = 0.f;
  if (use16) {
    const unsigned short* mp = m16 + row;
#pragma unroll
    for (int j = 0; j < 4; ++j) {
      int o = lane * 8 + j * 256;
      __builtin_prefetch(mp + o + 1024, 0, 0);
      v8u mv = *(const v8u*)(mp + o);
      v4f q0 = *(const v4f*)(qp + o);
      v4f q1 = *(const v4f*)(qp + o + 4);
      acc += q0[0]*bf2f(mv[0]) + q0[1]*bf2f(mv[1]) + q0[2]*bf2f(mv[2]) + q0[3]*bf2f(mv[3]);
      acc += q1[0]*bf2f(mv[4]) + q1[1]*bf2f(mv[5]) + q1[2]*bf2f(mv[6]) + q1[3]*bf2f(mv[7]);
    }
  } else {
    const float* mp = m32 + row;
#pragma unroll
    for (int j = 0; j < 8; ++j) {
      int o = lane * 4 + j * 128;
      v4f mv = *(const v4f*)(mp + o);
      v4f qv = *(const v4f*)(qp + o);
      acc += mv[0]*qv[0] + mv[1]*qv[1] + mv[2]*qv[2] + mv[3]*qv[3];
    }
  }
#pragma unroll
  for (int m = 16; m >= 1; m >>= 1) acc += __shfl_xor(acc, m, 32);
  if (lane == 0) logits[(size_t)b * L_ + l] = acc;
}

// ---------------------------------------------------------------------------
// softmax over L + weighted sum: out[b,:] = sum_l softmax(logits)[l]*M[b,l,:]
// one block (256 thr) per batch; writes fp32 + f16 copies of the result.
// ---------------------------------------------------------------------------
__global__ void k_attn_combine(const unsigned short* __restrict__ m16,
                               const float* __restrict__ m32, int use16,
                               const float* __restrict__ logits,
                               float* __restrict__ outf,
                               _Float16* __restrict__ outh) {
  __shared__ float red[256];
  __shared__ float alpha[L_];
  int b = blockIdx.x, tid = threadIdx.x;
  float l0 = logits[(size_t)b * L_ + tid];
  float l1 = logits[(size_t)b * L_ + tid + 256];
  red[tid] = fmaxf(l0, l1);
  __syncthreads();
  for (int s = 128; s >= 1; s >>= 1) {
    if (tid < s) red[tid] = fmaxf(red[tid], red[tid + s]);
    __syncthreads();
  }
  float mx = red[0];
  __syncthreads();
  float e0 = __expf(l0 - mx), e1 = __expf(l1 - mx);
  alpha[tid] = e0; alpha[tid + 256] = e1;
  red[tid] = e0 + e1;
  __syncthreads();
  for (int s = 128; s >= 1; s >>= 1) {
    if (tid < s) red[tid] += red[tid + s];
    __syncthreads();
  }
  float inv = 1.f / red[0];
  alpha[tid] *= inv; alpha[tid + 256] *= inv;
  __syncthreads();

  int h0 = tid * 4;
  float a0 = 0.f, a1 = 0.f, a2 = 0.f, a3 = 0.f;
  if (use16) {
    const unsigned short* mp = m16 + (size_t)b * L_ * H2_ + h0;
    for (int l = 0; l < L_; ++l) {
      __builtin_prefetch(mp + (size_t)(l + 8) * H2_, 0, 0);
      float w = alpha[l];
      v4u mv = *(const v4u*)(mp + (size_t)l * H2_);
      a0 += w * bf2f(mv[0]); a1 += w * bf2f(mv[1]);
      a2 += w * bf2f(mv[2]); a3 += w * bf2f(mv[3]);
    }
  } else {
    const float* mp = m32 + (size_t)b * L_ * H2_ + h0;
    for (int l = 0; l < L_; ++l) {
      __builtin_prefetch(mp + (size_t)(l + 8) * H2_, 0, 0);
      float w = alpha[l];
      v4f mv = *(const v4f*)(mp + (size_t)l * H2_);
      a0 += w * mv[0]; a1 += w * mv[1]; a2 += w * mv[2]; a3 += w * mv[3];
    }
  }
  size_t o = (size_t)b * H2_ + h0;
  outf[o + 0] = a0; outf[o + 1] = a1; outf[o + 2] = a2; outf[o + 3] = a3;
  outh[o + 0] = (_Float16)a0; outh[o + 1] = (_Float16)a1;
  outh[o + 2] = (_Float16)a2; outh[o + 3] = (_Float16)a3;
}

// ---------------------------------------------------------------------------
// WMMA 16x16x32 f16 tile loads (wave32), contiguous k order, same for A and B.
// A: row-major [M,K]; lane holds row m0+(lane&15), 16 contiguous K
//    starting at k0+(lane>>4)*16.
// B^T: row-major [N,K]; lane holds col n0+(lane&15), 16 contiguous K
//    starting at k0+(lane>>4)*16.
// ---------------------------------------------------------------------------
__device__ __forceinline__ v16h load_a16(const _Float16* __restrict__ A,
                                         int lda, int m0, int k0, int lane) {
  int r    = m0 + (lane & 15);
  int koff = (lane >> 4) * 16;
  const _Float16* p = A + (size_t)r * lda + k0 + koff;
  return *(const v16h*)p;
}
__device__ __forceinline__ v16h load_b16(const _Float16* __restrict__ Bt,
                                         int ldb, int n0, int k0, int lane) {
  int c  = n0 + (lane & 15);
  int kk = k0 + (lane >> 4) * 16;
  return *(const v16h*)(Bt + (size_t)c * ldb + kk);
}

// tmp = s @ theta3 : C[M,N] = A[M,K] * B[K,N], B given transposed [N,K]
__global__ void k_gemm_nt(const _Float16* __restrict__ A,
                          const _Float16* __restrict__ Bt,
                          float* __restrict__ C, int M, int N, int K) {
  int gid  = blockIdx.x * blockDim.x + threadIdx.x;
  int wave = gid >> 5, lane = gid & 31;
  int nT = N >> 4;
  int m0 = (wave / nT) << 4;
  int n0 = (wave % nT) << 4;
  v8f c = {};
  for (int k0 = 0; k0 < K; k0 += 32) {
    v16h a  = load_a16(A, K, m0, k0, lane);
    v16h bv = load_b16(Bt, K, n0, k0, lane);
    c = __builtin_amdgcn_wmma_f32_16x16x32_f16(false, a, false, bv,
                                               (short)0, c, false, false);
  }
  int col = n0 + (lane & 15);
  int rb  = m0 + ((lane >> 4) << 3);
#pragma unroll
  for (int r = 0; r < 8; ++r) C[(size_t)(rb + r) * N + col] = c[r];
}

// GRU gate GEMMs (fused, dual WMMA chains):
//   GI[b,n] = sum_k x[b,k]*W_ih[n,k] + b_ih[n]
//   GH[b,n] = sum_k s[b,k]*W_hh[n,k] + b_hh[n]
__global__ void k_gemm_gru(const _Float16* __restrict__ X,
                           const _Float16* __restrict__ S,
                           const _Float16* __restrict__ Wih,
                           const _Float16* __restrict__ Whh,
                           const float* __restrict__ bih,
                           const float* __restrict__ bhh,
                           float* __restrict__ GI, float* __restrict__ GH) {
  const int K = H2_, N = N3_;
  int gid  = blockIdx.x * blockDim.x + threadIdx.x;
  int wave = gid >> 5, lane = gid & 31;
  int nT = N >> 4;
  int m0 = (wave / nT) << 4;
  int n0 = (wave % nT) << 4;
  v8f ci = {}, ch = {};
  for (int k0 = 0; k0 < K; k0 += 32) {
    v16h ax = load_a16(X, K, m0, k0, lane);
    v16h as = load_a16(S, K, m0, k0, lane);
    v16h bi = load_b16(Wih, K, n0, k0, lane);
    v16h bh = load_b16(Whh, K, n0, k0, lane);
    ci = __builtin_amdgcn_wmma_f32_16x16x32_f16(false, ax, false, bi,
                                                (short)0, ci, false, false);
    ch = __builtin_amdgcn_wmma_f32_16x16x32_f16(false, as, false, bh,
                                                (short)0, ch, false, false);
  }
  int col = n0 + (lane & 15);
  int rb  = m0 + ((lane >> 4) << 3);
  float bi_c = bih[col], bh_c = bhh[col];
#pragma unroll
  for (int r = 0; r < 8; ++r) {
    GI[(size_t)(rb + r) * N + col] = ci[r] + bi_c;
    GH[(size_t)(rb + r) * N + col] = ch[r] + bh_c;
  }
}

// GRU pointwise: s' = (1-z)*n + z*s ; writes fp32 + f16 state
__global__ void k_gate(const float* __restrict__ GI, const float* __restrict__ GH,
                       float* __restrict__ S, _Float16* __restrict__ S16) {
  int idx = blockIdx.x * blockDim.x + threadIdx.x; // B*H2
  int b = idx >> 10, h = idx & (H2_ - 1);
  size_t base = (size_t)b * N3_;
  float r = 1.f / (1.f + __expf(-(GI[base + h] + GH[base + h])));
  float z = 1.f / (1.f + __expf(-(GI[base + H2_ + h] + GH[base + H2_ + h])));
  float n = tanhf(GI[base + 2 * H2_ + h] + r * GH[base + 2 * H2_ + h]);
  float sn = (1.f - z) * n + z * S[idx];
  S[idx] = sn;
  S16[idx] = (_Float16)sn;
}

// classifier partial: pacc[b,c] += feats(b) . W_cls[c,:]  (bias added at end)
__global__ void k_cls(const float* __restrict__ S, const float* __restrict__ X,
                      const float* __restrict__ Wc, float* __restrict__ pacc) {
  __shared__ float red[3 * 256];
  int b = blockIdx.x, tid = threadIdx.x;
  float p0 = 0.f, p1 = 0.f, p2 = 0.f;
  for (int h = tid; h < H2_; h += 256) {
    float s = S[(size_t)b * H2_ + h], x = X[(size_t)b * H2_ + h];
    float a = fabsf(s - x), m = s * x;
    p0 += Wc[h]        * s + Wc[1024 + h] * x + Wc[2048 + h]  * a + Wc[3072 + h]  * m;
    p1 += Wc[4096 + h] * s + Wc[5120 + h] * x + Wc[6144 + h]  * a + Wc[7168 + h]  * m;
    p2 += Wc[8192 + h] * s + Wc[9216 + h] * x + Wc[10240 + h] * a + Wc[11264 + h] * m;
  }
  red[tid] = p0; red[256 + tid] = p1; red[512 + tid] = p2;
  __syncthreads();
  for (int s2 = 128; s2 >= 1; s2 >>= 1) {
    if (tid < s2) {
      red[tid] += red[tid + s2];
      red[256 + tid] += red[256 + tid + s2];
      red[512 + tid] += red[512 + tid + s2];
    }
    __syncthreads();
  }
  if (tid == 0) {
    pacc[b * 3 + 0] += red[0];
    pacc[b * 3 + 1] += red[256];
    pacc[b * 3 + 2] += red[512];
  }
}

__global__ void k_final(const float* __restrict__ pacc,
                        const float* __restrict__ bcls,
                        float* __restrict__ out, float scale) {
  int i = blockIdx.x * blockDim.x + threadIdx.x;
  if (i < B_ * 3) out[i] = pacc[i] * scale + bcls[i % 3];
}

// ---------------------------------------------------------------------------
// Host launcher
// ---------------------------------------------------------------------------
extern "C" void kernel_launch(void* const* d_in, const int* in_sizes, int n_in,
                              void* d_out, int out_size, void* d_ws, size_t ws_size,
                              hipStream_t stream) {
  (void)in_sizes; (void)n_in; (void)out_size;
  const float* m_p   = (const float*)d_in[0];
  const float* m_h   = (const float*)d_in[1];
  const float* th2   = (const float*)d_in[2];
  const float* th3   = (const float*)d_in[3];
  const float* W_ih  = (const float*)d_in[4];
  const float* W_hh  = (const float*)d_in[5];
  const float* b_ih  = (const float*)d_in[6];
  const float* b_hh  = (const float*)d_in[7];
  const float* W_cls = (const float*)d_in[8];
  const float* b_cls = (const float*)d_in[9];
  float* out = (float*)d_out;

  // Workspace carve-up (all sizes 4KB-aligned)
  char* ws = (char*)d_ws;
  size_t off = 0;
  auto carve = [&](size_t bytes) { char* p = ws + off; off += bytes; return p; };
  _Float16* th3T  = (_Float16*)carve((size_t)H2_ * H2_ * 2);   // 2 MB
  _Float16* wih16 = (_Float16*)carve((size_t)N3_ * H2_ * 2);   // 6 MB
  _Float16* whh16 = (_Float16*)carve((size_t)N3_ * H2_ * 2);   // 6 MB
  float*    s32   = (float*)   carve((size_t)B_ * H2_ * 4);
  _Float16* s16   = (_Float16*)carve((size_t)B_ * H2_ * 2);
  float*    x32   = (float*)   carve((size_t)B_ * H2_ * 4);
  _Float16* x16   = (_Float16*)carve((size_t)B_ * H2_ * 2);
  float*    tmp32 = (float*)   carve((size_t)B_ * H2_ * 4);
  float*    GI    = (float*)   carve((size_t)B_ * N3_ * 4);    // 1.5 MB
  float*    GH    = (float*)   carve((size_t)B_ * N3_ * 4);
  float*    logit = (float*)   carve((size_t)B_ * L_ * 4);
  float*    pacc  = (float*)   carve(4096);
  const size_t SZ_MP16 = (size_t)B_ * L_ * H2_ * 2;            // 128 MB
  int use16 = (ws_size >= off + SZ_MP16) ? 1 : 0;
  unsigned short* mp16 = use16 ? (unsigned short*)carve(SZ_MP16) : nullptr;

  const int T = 256;
  // --- one-time precompute ---
  {
    int n = N3_ * H2_;
    k_f32_to_f16<<<(n + T - 1) / T, T, 0, stream>>>(W_ih, wih16, n);
    k_f32_to_f16<<<(n + T - 1) / T, T, 0, stream>>>(W_hh, whh16, n);
    k_transpose_f16<<<(H2_ * H2_) / T, T, 0, stream>>>(th3, th3T);
    if (use16) {
      int n4 = (B_ * L_ * H2_) / 4;
      k_f32_to_bf16x4<<<(n4 + T - 1) / T, T, 0, stream>>>(m_p, mp16, n4);
    }
    k_zero<<<2, T, 0, stream>>>(pacc, B_ * 3);
  }

  // --- s0 = softmax(theta2 . m_h) weighted sum of m_h ---
  const int DOT_BLOCKS = (B_ * L_ * 32) / T;  // 8192
  k_attn_dots<<<DOT_BLOCKS, T, 0, stream>>>(nullptr, m_h, 0, th2, 0, logit);
  k_attn_combine<<<B_, T, 0, stream>>>(nullptr, m_h, 0, logit, s32, s16);

  // --- 8 decode steps ---
  for (int step = 0; step < STEP_; ++step) {
    // tmp = s @ theta3   (128x1024x1024, WMMA): 8*64 = 512 wave-tiles
    k_gemm_nt<<<(512 * 32) / T, T, 0, stream>>>(s16, th3T, tmp32, B_, H2_, H2_);
    // beta attention over m_p -> x
    k_attn_dots<<<DOT_BLOCKS, T, 0, stream>>>(mp16, m_p, use16, tmp32, H2_, logit);
    k_attn_combine<<<B_, T, 0, stream>>>(mp16, m_p, use16, logit, x32, x16);
    // GRU gate GEMMs (128x1024x3072 twice, dual WMMA): 8*192 = 1536 wave-tiles
    k_gemm_gru<<<(1536 * 32) / T, T, 0, stream>>>(x16, s16, wih16, whh16,
                                                  b_ih, b_hh, GI, GH);
    // pointwise GRU update (writes s32/s16 for next step)
    k_gate<<<(B_ * H2_) / T, T, 0, stream>>>(GI, GH, s32, s16);
    // classifier accumulation for included steps
    if ((INCLUDED_MASK >> step) & 1)
      k_cls<<<B_, T, 0, stream>>>(s32, x32, W_cls, pacc);
  }

  k_final<<<2, T, 0, stream>>>(pacc, b_cls, out, 1.0f / (float)INCLUDED_CNT);
}
